// PosteriorLinearizationMarkov_15479062134818
// MI455X (gfx1250) — compile-verified, matrix operand order checked by reading, not run on hardware
//
#include <hip/hip_runtime.h>
#include <hip/hip_bf16.h>
#include <math.h>

// CDNA5 WMMA vector types (wave32)
typedef __attribute__((ext_vector_type(16))) _Float16 v16h;
typedef __attribute__((ext_vector_type(8)))  float    v8f;

// TDM descriptor operand types (per probe: g0=uint32x4, g1=int32x8, g2/g3=int32x4)
typedef unsigned int u32x4 __attribute__((ext_vector_type(4)));
typedef int          i32x4 __attribute__((ext_vector_type(4)));
typedef int          i32x8 __attribute__((ext_vector_type(8)));

#define PREC_FLOOR 1e-6f
#define DAMP       0.5f
#define MAX_ITER   4
#define CHUNK      512

// ---------------------------------------------------------------------------
// TDM: 1-D contiguous tile (tensor == tile), 4-byte elements, no multicast,
// no padding, no iteration.  D# bitfields per CDNA5 ISA ch.8.
// ---------------------------------------------------------------------------
__device__ __forceinline__ void tdm_load_1d(const float* gsrc, unsigned lds_off,
                                            unsigned nelem)
{
  unsigned long long ga = (unsigned long long)(uintptr_t)gsrc;
  u32x4 g0;
  g0[0] = 1u;                                   // count=1 (valid user D#)
  g0[1] = lds_off;                              // lds_addr (bytes)
  g0[2] = (unsigned)(ga & 0xFFFFFFFFu);         // global_addr[31:0]
  g0[3] = (unsigned)((ga >> 32) & 0x01FFFFFFu)  // global_addr[56:32]
          | 0x80000000u;                        // type=2 ("image")
  i32x8 g1;
  g1[0] = (int)(2u << 16);                      // data_size=2 (4 bytes)
  g1[1] = (int)((nelem & 0xFFFFu) << 16);       // tensor_dim0[15:0]
  g1[2] = (int)(((nelem >> 16) & 0xFFFFu) | (1u << 16)); // td0 hi | tensor_dim1=1
  g1[3] = (int)((nelem & 0xFFFFu) << 16);       // tile_dim0 = nelem
  g1[4] = 0;                                    // tile_dim1=0, tile_dim2=0
  g1[5] = (int)nelem;                           // tensor_dim0_stride lo32
  g1[6] = 0;
  g1[7] = 0;
  i32x4 gz = {0, 0, 0, 0};
#if defined(__clang_major__) && (__clang_major__ >= 23)
  i32x8 gz8 = {0, 0, 0, 0, 0, 0, 0, 0};
  __builtin_amdgcn_tensor_load_to_lds(g0, g1, gz, gz, gz8, 0);
#else
  __builtin_amdgcn_tensor_load_to_lds(g0, g1, gz, gz, 0);
#endif
}

__device__ __forceinline__ unsigned lds_off_of(const void* p)
{
  // Flat LDS aperture: addr[31:0] is the LDS byte offset.
  return (unsigned)(uintptr_t)p;
}

// ---------------------------------------------------------------------------
// Kernel 0: discretise the Matern-3/2 SSM (A_n, Q_n) and init EP state.
// ---------------------------------------------------------------------------
__global__ __launch_bounds__(256) void pl_precompute(
    const float* __restrict__ times, const float* __restrict__ log_ell,
    const float* __restrict__ log_var,
    float* __restrict__ A4, float* __restrict__ Q3,
    float* __restrict__ nat1, float* __restrict__ nat2,
    float* __restrict__ qm, float* __restrict__ qv, int N)
{
  int n = blockIdx.x * blockDim.x + threadIdx.x;
  if (n >= N) return;
  float ell    = __expf(log_ell[0]);
  float sigma2 = __expf(log_var[0]);
  float lam    = 1.7320508075688772f / ell;       // sqrt(3)/ell
  float dt = (n == 0) ? 0.0f : (times[n] - times[n - 1]);
  float ld = lam * dt;
  float e  = __expf(-ld);
  float a00 =  e * (1.0f + ld);
  float a01 =  e * dt;
  float a10 = -e * lam * lam * dt;
  float a11 =  e * (1.0f - ld);
  float s  = sigma2;
  float s2 = lam * lam * sigma2;
  float m00 = a00 * s, m01 = a01 * s2, m10 = a10 * s, m11 = a11 * s2;
  float q00 = s  - (m00 * a00 + m01 * a01);
  float q01 =      -(m00 * a10 + m01 * a11);
  float q11 = s2 - (m10 * a10 + m11 * a11);
  A4[4 * n + 0] = a00; A4[4 * n + 1] = a01;
  A4[4 * n + 2] = a10; A4[4 * n + 3] = a11;
  Q3[3 * n + 0] = q00; Q3[3 * n + 1] = q01; Q3[3 * n + 2] = q11;
  nat1[n] = 0.0f; nat2[n] = PREC_FLOOR;
  qm[n] = 0.0f;   qv[n] = 1.0f;
}

// ---------------------------------------------------------------------------
// Kernel 1: EP site update via v_wmma_f32_16x16x32_f16 quadrature contraction.
// ---------------------------------------------------------------------------
__global__ __launch_bounds__(256) void pl_site_update(
    const int* __restrict__ y, const float* __restrict__ qm,
    const float* __restrict__ qv,
    float* __restrict__ nat1, float* __restrict__ nat2, int N)
{
  constexpr float GX[32] = {
    -7.6190485f, -6.5105903f, -5.5787401f, -4.7345833f, -3.9439673f,
    -3.1890148f, -2.4586637f, -1.7452473f, -1.0429459f, -0.3469642f,
     0.3469642f,  1.0429459f,  1.7452473f,  2.4586637f,  3.1890148f,
     3.9439673f,  4.7345833f,  5.5787401f,  6.5105903f,  7.6190485f,
     0.0f, 0.0f, 0.0f, 0.0f, 0.0f, 0.0f, 0.0f, 0.0f, 0.0f, 0.0f, 0.0f, 0.0f };
  constexpr float GW[32] = {
    1.2578006e-13f, 2.4820644e-10f, 6.1275036e-08f, 4.4021241e-06f,
    1.2882628e-04f, 1.8301031e-03f, 1.3997837e-02f, 6.1506339e-02f,
    1.6173933e-01f, 2.6079306e-01f, 2.6079306e-01f, 1.6173933e-01f,
    6.1506339e-02f, 1.3997837e-02f, 1.8301031e-03f, 1.2882628e-04f,
    4.4021241e-06f, 6.1275036e-08f, 2.4820644e-10f, 1.2578006e-13f,
    0.0f, 0.0f, 0.0f, 0.0f, 0.0f, 0.0f, 0.0f, 0.0f, 0.0f, 0.0f, 0.0f, 0.0f };

  int lane = threadIdx.x & 31;
  int wave = (int)((blockIdx.x * blockDim.x + threadIdx.x) >> 5);
  int base = wave * 16;
  if (base >= N) return;                 // uniform per wave: EXEC stays all-1s
  int n = base + (lane & 15);

  float qvn = qv[n], qmn = qm[n];
  float n1 = nat1[n], n2 = nat2[n];
  float iqv = 1.0f / qvn;
  float cav_prec = fmaxf(iqv - n2, PREC_FLOOR);
  float cav_var  = 1.0f / cav_prec;
  float cav_mean = cav_var * (qmn * iqv - n1);
  float sdev = sqrtf(cav_var);

  const bool lo = (lane < 16);
  v16h b;
  #pragma unroll
  for (int j = 0; j < 16; ++j) {
    float xk = lo ? GX[j] : GX[j + 16];
    float wk = lo ? GW[j] : GW[j + 16];
    float arg = fminf(cav_mean + sdev * xk, 30.0f);
    float val = fminf(wk * __expf(arg), 60000.0f);
    b[j] = (_Float16)val;
  }
  v16h a;
  #pragma unroll
  for (int j = 0; j < 16; ++j) a[j] = (_Float16)1.0f;

  v8f c = {};
  c = __builtin_amdgcn_wmma_f32_16x16x32_f16(
        false, a, false, b, (short)0, c, false, false);

  if (lo) {
    float E = c[0];
    float yf = (float)y[n];
    float new_prec = fmaxf(E, PREC_FLOOR);
    float new_nat1 = (yf - E) + new_prec * cav_mean;
    nat1[n] = (1.0f - DAMP) * n1 + DAMP * new_nat1;
    nat2[n] = fmaxf((1.0f - DAMP) * n2 + DAMP * new_prec, PREC_FLOOR);
  }
}

// ---------------------------------------------------------------------------
// Kernel 2: sequential Kalman filter. TDM double-buffers chunk inputs into
// LDS (tensor_load_to_lds / s_wait_tensorcnt); serial lane reads LDS via
// ds_load (direct shared-array access keeps addrspace(3) provenance).
// ---------------------------------------------------------------------------
__global__ __launch_bounds__(32) void pl_forward(
    const float* __restrict__ A4, const float* __restrict__ Q3,
    const float* __restrict__ nat1, const float* __restrict__ nat2,
    const float* __restrict__ log_ell, const float* __restrict__ log_var,
    float* __restrict__ mp, float* __restrict__ Pp,
    float* __restrict__ mf, float* __restrict__ Pf, int N)
{
  __shared__ __align__(16) float sA[2][CHUNK * 4];
  __shared__ __align__(16) float sQ[2][CHUNK * 3];
  __shared__ __align__(16) float s1[2][CHUNK];
  __shared__ __align__(16) float s2[2][CHUNK];

  const int lane = threadIdx.x;
  const int nck = (N + CHUNK - 1) / CHUNK;

  // issue chunk 0
  {
    int cnt = (CHUNK < N) ? CHUNK : N;
    tdm_load_1d(A4, lds_off_of(&sA[0][0]), 4u * (unsigned)cnt);
    tdm_load_1d(Q3, lds_off_of(&sQ[0][0]), 3u * (unsigned)cnt);
    tdm_load_1d(nat1, lds_off_of(&s1[0][0]), (unsigned)cnt);
    tdm_load_1d(nat2, lds_off_of(&s2[0][0]), (unsigned)cnt);
  }

  float ell    = __expf(log_ell[0]);
  float sigma2 = __expf(log_var[0]);
  float lam    = 1.7320508075688772f / ell;
  float m0 = 0.0f, m1 = 0.0f;
  float p00 = sigma2, p01 = 0.0f, p11 = lam * lam * sigma2;  // P_inf

  for (int c = 0; c < nck; ++c) {
    if (c + 1 < nck) {                      // prefetch next chunk into other buf
      int s = (c + 1) * CHUNK;
      int cnt = (s + CHUNK < N) ? CHUNK : (N - s);
      int b = (c + 1) & 1;
      tdm_load_1d(A4 + 4 * s, lds_off_of(&sA[b][0]), 4u * (unsigned)cnt);
      tdm_load_1d(Q3 + 3 * s, lds_off_of(&sQ[b][0]), 3u * (unsigned)cnt);
      tdm_load_1d(nat1 + s, lds_off_of(&s1[b][0]), (unsigned)cnt);
      tdm_load_1d(nat2 + s, lds_off_of(&s2[b][0]), (unsigned)cnt);
      // TDM completes in issue order: <=4 outstanding means chunk c landed.
      __builtin_amdgcn_s_wait_tensorcnt(4);
    } else {
      __builtin_amdgcn_s_wait_tensorcnt(0);
    }
    asm volatile("" ::: "memory");

    if (lane == 0) {
      int s = c * CHUNK;
      int cnt = (s + CHUNK < N) ? CHUNK : (N - s);
      int b = c & 1;
      const float4* La4 = (const float4*)&sA[b][0];   // ds_load_b128 rows
      for (int i = 0; i < cnt; ++i) {
        int n = s + i;
        float4 av = La4[i];
        float a00 = av.x, a01 = av.y, a10 = av.z, a11 = av.w;
        float q00 = sQ[b][3*i], q01 = sQ[b][3*i+1], q11 = sQ[b][3*i+2];
        float mp0 = a00 * m0 + a01 * m1;
        float mp1 = a10 * m0 + a11 * m1;
        float t00 = a00 * p00 + a01 * p01;
        float t01 = a00 * p01 + a01 * p11;
        float t10 = a10 * p00 + a11 * p01;
        float t11 = a10 * p01 + a11 * p11;
        float pp00 = t00 * a00 + t01 * a01 + q00;
        float pp01 = t00 * a10 + t01 * a11 + q01;
        float pp11 = t10 * a10 + t11 * a11 + q11;
        float R  = 1.0f / s2[b][i];
        float r  = s1[b][i] * R;
        float v  = r - mp0;
        float S  = pp00 + R;
        float iS = 1.0f / S;
        float K0 = pp00 * iS, K1 = pp01 * iS;
        m0 = mp0 + v * K0;
        m1 = mp1 + v * K1;
        p00 = pp00 - pp00 * pp00 * iS;
        p01 = pp01 - pp00 * pp01 * iS;
        p11 = pp11 - pp01 * pp01 * iS;
        ((float2*)mp)[n] = make_float2(mp0, mp1);     // global_store_b64
        Pp[3*n] = pp00; Pp[3*n+1] = pp01; Pp[3*n+2] = pp11;
        ((float2*)mf)[n] = make_float2(m0, m1);
        Pf[3*n] = p00; Pf[3*n+1] = p01; Pf[3*n+2] = p11;
      }
    }
    asm volatile("" ::: "memory");
  }
}

// ---------------------------------------------------------------------------
// Kernel 3: RTS smoother (reverse). Same TDM double-buffering, 5 streams.
// For chunk [cs, ce): A/m_pred/P_pred staged from row cs+1, m_filt/P_filt
// from row cs; within-chunk index i = n - cs addresses both.
// ---------------------------------------------------------------------------
__global__ __launch_bounds__(32) void pl_backward(
    const float* __restrict__ A4,
    const float* __restrict__ mp, const float* __restrict__ Pp,
    const float* __restrict__ mf, const float* __restrict__ Pf,
    float* __restrict__ qm, float* __restrict__ qv, int N)
{
  __shared__ __align__(16) float sA [2][CHUNK * 4];
  __shared__ __align__(16) float sMP[2][CHUNK * 2];
  __shared__ __align__(16) float sPP[2][CHUNK * 3];
  __shared__ __align__(16) float sMF[2][CHUNK * 2];
  __shared__ __align__(16) float sPF[2][CHUNK * 3];

  const int lane = threadIdx.x;
  const int M = N - 1;                        // number of smoother steps
  const int nck = (M + CHUNK - 1) / CHUNK;

  // issue top chunk
  {
    int c = nck - 1, b = c & 1;
    int cs = c * CHUNK;
    int cnt = M - cs;                         // <= CHUNK
    tdm_load_1d(A4 + 4 * (cs + 1), lds_off_of(&sA [b][0]), 4u * (unsigned)cnt);
    tdm_load_1d(mp + 2 * (cs + 1), lds_off_of(&sMP[b][0]), 2u * (unsigned)cnt);
    tdm_load_1d(Pp + 3 * (cs + 1), lds_off_of(&sPP[b][0]), 3u * (unsigned)cnt);
    tdm_load_1d(mf + 2 * cs,       lds_off_of(&sMF[b][0]), 2u * (unsigned)cnt);
    tdm_load_1d(Pf + 3 * cs,       lds_off_of(&sPF[b][0]), 3u * (unsigned)cnt);
  }

  float ms0 = 0.f, ms1 = 0.f, ps00 = 0.f, ps01 = 0.f, ps11 = 0.f;
  if (lane == 0) {
    ms0 = mf[2 * (N - 1)]; ms1 = mf[2 * (N - 1) + 1];
    ps00 = Pf[3 * (N - 1)]; ps01 = Pf[3 * (N - 1) + 1]; ps11 = Pf[3 * (N - 1) + 2];
    qm[N - 1] = ms0;
    qv[N - 1] = fmaxf(ps00, 1e-12f);
  }

  for (int c = nck - 1; c >= 0; --c) {
    if (c > 0) {                              // prefetch next (lower) chunk
      int cp = c - 1, b = cp & 1;
      int cs = cp * CHUNK;
      int cnt = CHUNK;                        // interior chunks are full
      tdm_load_1d(A4 + 4 * (cs + 1), lds_off_of(&sA [b][0]), 4u * (unsigned)cnt);
      tdm_load_1d(mp + 2 * (cs + 1), lds_off_of(&sMP[b][0]), 2u * (unsigned)cnt);
      tdm_load_1d(Pp + 3 * (cs + 1), lds_off_of(&sPP[b][0]), 3u * (unsigned)cnt);
      tdm_load_1d(mf + 2 * cs,       lds_off_of(&sMF[b][0]), 2u * (unsigned)cnt);
      tdm_load_1d(Pf + 3 * cs,       lds_off_of(&sPF[b][0]), 3u * (unsigned)cnt);
      // chunk c's 5 descriptors are done once only chunk c-1's 5 remain.
      __builtin_amdgcn_s_wait_tensorcnt(5);
    } else {
      __builtin_amdgcn_s_wait_tensorcnt(0);
    }
    asm volatile("" ::: "memory");

    if (lane == 0) {
      int cs = c * CHUNK;
      int ce = (cs + CHUNK < M) ? (cs + CHUNK) : M;
      int b = c & 1;
      const float4* La4  = (const float4*)&sA [b][0];  // ds_load_b128
      const float2* Lmp2 = (const float2*)&sMP[b][0];  // ds_load_b64
      const float2* Lmf2 = (const float2*)&sMF[b][0];
      for (int n = ce - 1; n >= cs; --n) {
        int i = n - cs;
        float4 av = La4[i];
        float a00 = av.x, a01 = av.y, a10 = av.z, a11 = av.w;
        float2 mpv = Lmp2[i];
        float mp0 = mpv.x, mp1 = mpv.y;
        float pp00 = sPP[b][3*i], pp01 = sPP[b][3*i+1], pp11 = sPP[b][3*i+2];
        float2 mfv = Lmf2[i];
        float fm0 = mfv.x, fm1 = mfv.y;
        float f00 = sPF[b][3*i], f01 = sPF[b][3*i+1], f11 = sPF[b][3*i+2];
        // B = A * P_f
        float b00 = a00 * f00 + a01 * f01;
        float b01 = a00 * f01 + a01 * f11;
        float b10 = a10 * f00 + a11 * f01;
        float b11 = a10 * f01 + a11 * f11;
        // G = (P_pred^{-1} B)^T
        float det = pp00 * pp11 - pp01 * pp01;
        float id  = 1.0f / det;
        float i00 = pp11 * id, i01 = -pp01 * id, i11 = pp00 * id;
        float ib00 = i00 * b00 + i01 * b10;
        float ib01 = i00 * b01 + i01 * b11;
        float ib10 = i01 * b00 + i11 * b10;
        float ib11 = i01 * b01 + i11 * b11;
        float g00 = ib00, g01 = ib10, g10 = ib01, g11 = ib11;
        float dm0 = ms0 - mp0, dm1 = ms1 - mp1;
        float nm0 = fm0 + g00 * dm0 + g01 * dm1;
        float nm1 = fm1 + g10 * dm0 + g11 * dm1;
        float d00 = ps00 - pp00, d01 = ps01 - pp01, d11 = ps11 - pp11;
        float u00 = g00 * d00 + g01 * d01;
        float u01 = g00 * d01 + g01 * d11;
        float u10 = g10 * d00 + g11 * d01;
        float u11 = g10 * d01 + g11 * d11;
        float np00 = f00 + u00 * g00 + u01 * g01;
        float np01 = f01 + u00 * g10 + u01 * g11;
        float np11 = f11 + u10 * g10 + u11 * g11;
        ms0 = nm0; ms1 = nm1; ps00 = np00; ps01 = np01; ps11 = np11;
        qm[n] = ms0;
        qv[n] = fmaxf(ps00, 1e-12f);
      }
    }
    asm volatile("" ::: "memory");
  }
}

// ---------------------------------------------------------------------------
extern "C" void kernel_launch(void* const* d_in, const int* in_sizes, int n_in,
                              void* d_out, int out_size, void* d_ws, size_t ws_size,
                              hipStream_t stream)
{
  const float* times = (const float*)d_in[0];
  const int*   y     = (const int*)d_in[1];
  const float* lls   = (const float*)d_in[2];
  const float* lvs   = (const float*)d_in[3];
  const int N = in_sizes[0];
  (void)n_in; (void)out_size; (void)ws_size;

  float* ws = (float*)d_ws;
  size_t off = 0;
  float* A4   = ws + off; off += 4 * (size_t)N;
  float* Q3   = ws + off; off += 3 * (size_t)N;
  float* nat1 = ws + off; off += (size_t)N;
  float* nat2 = ws + off; off += (size_t)N;
  float* qm   = ws + off; off += (size_t)N;
  float* qv   = ws + off; off += (size_t)N;
  float* mp   = ws + off; off += 2 * (size_t)N;
  float* Pp   = ws + off; off += 3 * (size_t)N;
  float* mf   = ws + off; off += 2 * (size_t)N;
  float* Pf   = ws + off; off += 3 * (size_t)N;

  pl_precompute<<<(N + 255) / 256, 256, 0, stream>>>(
      times, lls, lvs, A4, Q3, nat1, nat2, qm, qv, N);

  for (int it = 0; it < MAX_ITER; ++it) {
    pl_site_update<<<(N + 127) / 128, 256, 0, stream>>>(y, qm, qv, nat1, nat2, N);
    pl_forward<<<1, 32, 0, stream>>>(A4, Q3, nat1, nat2, lls, lvs, mp, Pp, mf, Pf, N);
    float* qm_dst = (it == MAX_ITER - 1) ? (float*)d_out : qm;
    pl_backward<<<1, 32, 0, stream>>>(A4, mp, Pp, mf, Pf, qm_dst, qv, N);
  }
}
